// NPOSRegLoss_29592324669625
// MI455X (gfx1250) — matrix operand order repaired and checked by reading, not compiled, not run on hardware
//
#include <hip/hip_runtime.h>
#include <hip/hip_bf16.h>
#include <math.h>

#define ALPHA_C 0.1f
#define SIGMA_C 0.5f
#define KNN_K   50
#define NPTS    10
#define BATCH   8192
#define DIM     512
#define NBINS   256
#define TROWS   16

typedef __attribute__((ext_vector_type(4)))  _Float16 v4h;
typedef __attribute__((ext_vector_type(8)))  _Float16 v8h;
typedef __attribute__((ext_vector_type(16))) _Float16 v16h;
typedef __attribute__((ext_vector_type(8)))  float    v8f;

__device__ __forceinline__ float softplusf(float x) {
    // numerically stable log(1+exp(x))
    return fmaxf(x, 0.0f) + log1pf(expf(-fabsf(x)));
}

// ---------------------------------------------------------------------------
// Kernel 1: per-row L2 normalize -> f16 Zh; id logit = z.phi_w + phi_b;
// accumulate sum softplus(-logit) into acc.
// grid = BATCH blocks of 128 threads (4 elems/thread).
// ---------------------------------------------------------------------------
__global__ void normalize_idloss_kernel(const float* __restrict__ emb,
                                        const float* __restrict__ phi_w,
                                        const float* __restrict__ phi_b,
                                        _Float16* __restrict__ Zh,
                                        float* __restrict__ acc)
{
    const int row = blockIdx.x;
    const int t   = threadIdx.x;           // 128 threads
    __shared__ float red[128];

    float4 x = ((const float4*)(emb + (size_t)row * DIM))[t];
    red[t] = x.x*x.x + x.y*x.y + x.z*x.z + x.w*x.w;
    __syncthreads();
    for (int s = 64; s > 0; s >>= 1) { if (t < s) red[t] += red[t + s]; __syncthreads(); }
    const float rn = rsqrtf(red[0]);
    __syncthreads();

    const float zx = x.x*rn, zy = x.y*rn, zz = x.z*rn, zw = x.w*rn;
    v4h h; h[0] = (_Float16)zx; h[1] = (_Float16)zy; h[2] = (_Float16)zz; h[3] = (_Float16)zw;
    ((v4h*)(Zh + (size_t)row * DIM))[t] = h;

    float4 w = ((const float4*)phi_w)[t];
    red[t] = zx*w.x + zy*w.y + zz*w.z + zw*w.w;
    __syncthreads();
    for (int s = 64; s > 0; s >>= 1) { if (t < s) red[t] += red[t + s]; __syncthreads(); }
    if (t == 0) {
        float logit = red[0] + phi_b[0];
        atomicAdd(acc, softplusf(-logit));   // loss_id accumulates softplus(-x)
    }
}

// ---------------------------------------------------------------------------
// Kernel 2/4: WMMA sweep sim = Zh @ Zh^T for a 16-row tile, histogramming each
// row's similarities. PASS 1: coarse 256-bin histogram over sim in [-1,1].
// PASS 2: 256 fine sub-bins inside the row's selected coarse bin.
// Both passes compute bit-identical sims (same code path) -> consistent bins.
// 256 threads = 8 waves; wave w sweeps a PAIR of column tiles per iteration
// (two accumulators share each A fragment -> half the LDS traffic, 2x WMMA
// density). Diagonal excluded branchlessly via sentinel value -2.0 -> bin 0,
// which the top-down rank scan never reaches for K=50.
// ---------------------------------------------------------------------------
template<int PASS>
__global__ __launch_bounds__(256)
void simhist_kernel(const _Float16* __restrict__ Zh,
                    unsigned int* __restrict__ hist,
                    const int* __restrict__ rowBin)
{
    __shared__ __align__(16) _Float16 As[TROWS * DIM];   // 16 KB staged A rows
    __shared__ unsigned int lh[2][TROWS * NBINS];        // 32 KB (2 copies vs contention)

    const int r0   = blockIdx.x * TROWS;
    const int tid  = threadIdx.x;
    const int wave = tid >> 5;
    const int lane = tid & 31;
    const int lm   = lane & 15;       // A: row M, B: col N, C: col N
    const int hi   = lane >> 4;       // lane-half selector

    for (int i = tid; i < 2 * TROWS * NBINS; i += 256) ((unsigned int*)lh)[i] = 0u;
    for (int i = tid; i < TROWS * DIM / 8; i += 256)
        ((v8h*)As)[i] = ((const v8h*)(Zh + (size_t)r0 * DIM))[i];
    __syncthreads();

    unsigned int* myh = lh[wave & 1];

    int rb[8];
    if constexpr (PASS == 2) {
#pragma unroll
        for (int v = 0; v < 8; ++v) rb[v] = rowBin[r0 + v + 8 * hi];
    }

    // A operand as loaded: lane holds row M=lm, contiguous K chunk 16*hi (same as B)
    const int ksel = (hi == 0) ? 0 : 8;

    // branchless per-row histogram update for one 16x16 C tile
    auto hist_update = [&](const v8f& c, int col) {
#pragma unroll
        for (int v = 0; v < 8; ++v) {
            const int lrow = v + 8 * hi;        // C/D layout: M = v + 8*(lane>>4)
            // diagonal -> sentinel -2.0 -> bin 0 (never reached by rank scan)
            const float sv = (r0 + lrow == col) ? -2.0f : c[v];
            const float scaled = (sv + 1.0f) * 128.0f;
            int bin = (int)scaled;
            bin = bin < 0 ? 0 : (bin > 255 ? 255 : bin);
            if constexpr (PASS == 1) {
                atomicAdd(&myh[lrow * NBINS + bin], 1u);
            } else {
                if (bin == rb[v]) {             // fires for ~1/256 of elements
                    int fb = (int)((scaled - (float)bin) * 256.0f);
                    fb = fb < 0 ? 0 : (fb > 255 ? 255 : fb);
                    atomicAdd(&myh[lrow * NBINS + fb], 1u);
                }
            }
        }
    };

    for (int c0 = wave * 32; c0 < BATCH; c0 += 8 * 32) {
        if (c0 + 256 < BATCH) {
            __builtin_prefetch(Zh + (size_t)(c0 + 256 + lm) * DIM, 0, 1);
            __builtin_prefetch(Zh + (size_t)(c0 + 272 + lm) * DIM, 0, 1);
        }

        v8f cA = {0.f, 0.f, 0.f, 0.f, 0.f, 0.f, 0.f, 0.f};
        v8f cB = {0.f, 0.f, 0.f, 0.f, 0.f, 0.f, 0.f, 0.f};
        const _Float16* arow  = As + lm * DIM;
        const _Float16* brow0 = Zh + (size_t)(c0 + lm) * DIM;        // tile 0 cols
        const _Float16* brow1 = Zh + (size_t)(c0 + 16 + lm) * DIM;   // tile 1 cols
#pragma unroll 4
        for (int kb = 0; kb < DIM; kb += 32) {
            v8h a0 = *(const v8h*)(arow + kb + 16 * hi);
            v8h a1 = *(const v8h*)(arow + kb + 16 * hi + 8);
            v16h a = __builtin_shufflevector(a0, a1, 0,1,2,3,4,5,6,7,8,9,10,11,12,13,14,15);
            // B f16 32x16 layout: lane holds col N=lm, contiguous K chunk 16*hi
            v8h b00 = *(const v8h*)(brow0 + kb + 16 * hi);
            v8h b01 = *(const v8h*)(brow0 + kb + 16 * hi + 8);
            v8h b10 = *(const v8h*)(brow1 + kb + 16 * hi);
            v8h b11 = *(const v8h*)(brow1 + kb + 16 * hi + 8);
            v16h b0 = __builtin_shufflevector(b00, b01, 0,1,2,3,4,5,6,7,8,9,10,11,12,13,14,15);
            v16h b1 = __builtin_shufflevector(b10, b11, 0,1,2,3,4,5,6,7,8,9,10,11,12,13,14,15);
            cA = __builtin_amdgcn_wmma_f32_16x16x32_f16(false, a, false, b0,
                                                        (short)0, cA, false, false);
            cB = __builtin_amdgcn_wmma_f32_16x16x32_f16(false, a, false, b1,
                                                        (short)0, cB, false, false);
        }

        hist_update(cA, c0 + lm);
        hist_update(cB, c0 + 16 + lm);
    }
    __syncthreads();
    for (int i = tid; i < TROWS * NBINS; i += 256)
        hist[(size_t)r0 * NBINS + i] = lh[0][i] + lh[1][i];
}

// ---------------------------------------------------------------------------
// Kernel 3: per-row coarse scan. Find bin holding the K-th largest sim and the
// residual rank inside it.
// ---------------------------------------------------------------------------
__global__ void scan_coarse_kernel(const unsigned int* __restrict__ hist,
                                   int* __restrict__ rowBin,
                                   int* __restrict__ rowRem)
{
    const int r = blockIdx.x * blockDim.x + threadIdx.x;
    if (r >= BATCH) return;
    const unsigned int* h = hist + (size_t)r * NBINS;
    int cum = 0, b = NBINS - 1;
    for (; b >= 0; --b) {
        int c = (int)h[b];
        if (cum + c >= KNN_K) break;
        cum += c;
    }
    if (b < 0) b = 0;
    rowBin[r] = b;
    rowRem[r] = KNN_K - cum;   // need rem-th largest inside coarse bin b
}

// ---------------------------------------------------------------------------
// Kernel 5: per-row fine scan -> k-NN distance.
// ---------------------------------------------------------------------------
__global__ void scan_fine_kernel(const unsigned int* __restrict__ hist,
                                 const int* __restrict__ rowBin,
                                 const int* __restrict__ rowRem,
                                 float* __restrict__ knn)
{
    const int r = blockIdx.x * blockDim.x + threadIdx.x;
    if (r >= BATCH) return;
    const unsigned int* h = hist + (size_t)r * NBINS;
    const int rem = rowRem[r];
    int cum = 0, f = NBINS - 1;
    for (; f >= 0; --f) {
        int c = (int)h[f];
        if (cum + c >= rem) break;
        cum += c;
    }
    if (f < 0) f = 0;
    // sim value at midpoint of fine bin (resolution 2/65536)
    float s = -1.0f + ((float)rowBin[r] + ((float)f + 0.5f) / 256.0f) * (2.0f / 256.0f);
    knn[r] = sqrtf(fmaxf(2.0f - 2.0f * s, 0.0f));
}

// ---------------------------------------------------------------------------
// Kernel 6: top-P (P=10) indices of knn dists. Single workgroup, 1024 threads,
// iterative argmax in LDS with lowest-index tie-break (matches top_k).
// ---------------------------------------------------------------------------
__global__ __launch_bounds__(1024)
void topp_kernel(const float* __restrict__ knn, int* __restrict__ bidx)
{
    __shared__ float vals[BATCH];     // 32 KB (320 KB LDS on CDNA5 WGP)
    __shared__ float sm[1024];
    __shared__ int   si[1024];
    const int t = threadIdx.x;
    for (int i = t; i < BATCH; i += 1024) vals[i] = knn[i];
    __syncthreads();
    for (int p = 0; p < NPTS; ++p) {
        float bv = -1e30f; int bi = 0x7fffffff;
        for (int i = t; i < BATCH; i += 1024) {
            float v = vals[i];
            if (v > bv) { bv = v; bi = i; }
        }
        sm[t] = bv; si[t] = bi;
        __syncthreads();
        for (int s = 512; s > 0; s >>= 1) {
            if (t < s) {
                if (sm[t + s] > sm[t] || (sm[t + s] == sm[t] && si[t + s] < si[t])) {
                    sm[t] = sm[t + s]; si[t] = si[t + s];
                }
            }
            __syncthreads();
        }
        if (t == 0) { bidx[p] = si[0]; vals[si[0]] = -1e30f; }
        __syncthreads();
    }
}

// ---------------------------------------------------------------------------
// Kernel 7: v = normalize(emb[idx]) + sigma*noise; ood logits; final scalar.
// One block, 512 threads (one dim per thread).
// ---------------------------------------------------------------------------
__global__ __launch_bounds__(512)
void final_kernel(const float* __restrict__ emb,
                  const float* __restrict__ noise,
                  const float* __restrict__ phi_w,
                  const float* __restrict__ phi_b,
                  const int* __restrict__ bidx,
                  const float* __restrict__ acc,
                  float* __restrict__ out)
{
    __shared__ float red[512];
    __shared__ float oodsum;
    const int t = threadIdx.x;
    if (t == 0) oodsum = 0.0f;
    __syncthreads();
    const float w = phi_w[t];
    for (int p = 0; p < NPTS; ++p) {
        const int row = bidx[p];
        const float x = emb[(size_t)row * DIM + t];
        red[t] = x * x;
        __syncthreads();
        for (int s = 256; s > 0; s >>= 1) { if (t < s) red[t] += red[t + s]; __syncthreads(); }
        const float rn = rsqrtf(red[0]);
        __syncthreads();
        const float v = x * rn + SIGMA_C * noise[p * DIM + t];
        red[t] = v * w;
        __syncthreads();
        for (int s = 256; s > 0; s >>= 1) { if (t < s) red[t] += red[t + s]; __syncthreads(); }
        if (t == 0) oodsum += softplusf(red[0] + phi_b[0]);
        __syncthreads();
    }
    if (t == 0)
        out[0] = ALPHA_C * (acc[0] / (float)BATCH + oodsum / (float)NPTS);
}

// ---------------------------------------------------------------------------
extern "C" void kernel_launch(void* const* d_in, const int* in_sizes, int n_in,
                              void* d_out, int out_size, void* d_ws, size_t ws_size,
                              hipStream_t stream)
{
    const float* emb   = (const float*)d_in[0];
    // d_in[1] = labels (unused by the loss)
    const float* noise = (const float*)d_in[2];
    const float* phi_w = (const float*)d_in[3];
    const float* phi_b = (const float*)d_in[4];
    float* out = (float*)d_out;

    // workspace layout (~16.9 MB total)
    char* ws = (char*)d_ws;
    _Float16* Zh       = (_Float16*)ws;                                   // 8 MiB
    unsigned int* hist = (unsigned int*)(ws + (size_t)8 * 1024 * 1024);   // 8 MiB
    char* p = ws + (size_t)16 * 1024 * 1024;
    int*   rowBin = (int*)p;    p += (size_t)BATCH * 4;
    int*   rowRem = (int*)p;    p += (size_t)BATCH * 4;
    float* knn    = (float*)p;  p += (size_t)BATCH * 4;
    int*   bidx   = (int*)p;    p += 64;
    float* acc    = (float*)p;

    hipMemsetAsync(acc, 0, sizeof(float), stream);

    normalize_idloss_kernel<<<BATCH, 128, 0, stream>>>(emb, phi_w, phi_b, Zh, acc);
    simhist_kernel<1><<<BATCH / TROWS, 256, 0, stream>>>(Zh, hist, nullptr);
    scan_coarse_kernel<<<BATCH / 256, 256, 0, stream>>>(hist, rowBin, rowRem);
    simhist_kernel<2><<<BATCH / TROWS, 256, 0, stream>>>(Zh, hist, rowBin);
    scan_fine_kernel<<<BATCH / 256, 256, 0, stream>>>(hist, rowBin, rowRem, knn);
    topp_kernel<<<1, 1024, 0, stream>>>(knn, bidx);
    final_kernel<<<1, 512, 0, stream>>>(emb, noise, phi_w, phi_b, bidx, acc, out);
}
